// simclr_79637283602623
// MI455X (gfx1250) — compile-verified, matrix operand order checked by reading, not run on hardware
//
#include <hip/hip_runtime.h>
#include <hip/hip_bf16.h>

// ---------------------------------------------------------------------------
// SimCLR-style contrastive loss, fused for MI455X (gfx1250, wave32, WMMA).
//
//   B = 8192 rows per view, D = 128, N = 16384, T = 0.2 (1/T = 5)
//   loss = log( sum_{i!=j, member} exp(out_i . out_j / T) ) - (1/B) sum_i dot_i/T
//
// Heavy part: 16384x16384x128 GEMM (~68.7 GFLOP) -> v_wmma_f32_16x16x32_f16,
// fused with mask+exp+sum epilogue (never materializes the NxN matrix).
// The normalized f16 matrix (4 MB) is stored pre-swizzled in the exact WMMA
// fragment layout so every fragment load is two coalesced global_load_b128.
// Scalar reduction uses block-level LDS reduce + 64 cacheline-spread atomic
// buckets to avoid serializing ~131K same-address f32 atomics at one L2 slice.
// ---------------------------------------------------------------------------

typedef _Float16 v16h __attribute__((ext_vector_type(16)));
typedef float    v8f  __attribute__((ext_vector_type(8)));

#define B_ROWS   8192
#define DIM      128
#define NROWS    16384
#define INV_T    5.0f
#define NBUCKET  64
#define BSTRIDE  32          // floats between buckets -> 128 B, one cacheline each

// Workspace layout (bytes):
//   [0      .. 8192)  : denom buckets (64 buckets, stride 32 floats)
//   [8192   .. 16384) : pos-logit buckets (64 buckets, stride 32 floats)
//   [16384  .. +4MB)  : swizzled f16 'out' matrix (NROWS*DIM halves)
//   then NROWS u32    : membership bitmasks
//   then NROWS i32    : hq (copy of hard_q)

__device__ __forceinline__ unsigned pack2(float a, float b) {
    _Float16 ha = (_Float16)a, hb = (_Float16)b;
    unsigned short ua, ub;
    __builtin_memcpy(&ua, &ha, 2);
    __builtin_memcpy(&ub, &hb, 2);
    return (unsigned)ua | ((unsigned)ub << 16);
}

// Swizzled layout: per (rowTile t, kChunk c) block of 512 halves:
//   chunk0 (256 halves): lane L = g*16+m holds row (t*16+m), K = c*32 + g*8 .. +7
//   chunk1 (256 halves): lane L = g*16+m holds row (t*16+m), K = c*32 + 16 + g*8 .. +7
// => a WMMA 16x16x32 f16 fragment load is exactly uint4[lane] + uint4[32+lane].
__device__ __forceinline__ void store_row_swizzled(unsigned short* __restrict__ swz,
                                                   int row, int lane,
                                                   float4 v, float scale) {
    const int t = row >> 4, m = row & 15;
    const int k0 = lane * 4;            // 4 consecutive K per lane
    const int c = k0 >> 5;
    const int kk = k0 & 31;
    const int g     = (kk >> 3) & 1;
    const int chunk = (kk >> 4) & 1;
    const int pos   = kk & 7;           // 0 or 4 -> 8-byte aligned store
    const size_t idx = ((size_t)(t * 4 + c) << 9) + (size_t)chunk * 256
                     + (size_t)(g * 16 + m) * 8 + pos;
    unsigned long long p =
        (unsigned long long)pack2(v.x * scale, v.y * scale) |
        ((unsigned long long)pack2(v.z * scale, v.w * scale) << 32);
    *(unsigned long long*)(swz + idx) = p;
}

// One wave per row index i: normalizes x[i] and x_aug[i], writes both rows
// (i and B+i) of the swizzled f16 matrix, accumulates pos logit sum.
__global__ __launch_bounds__(256) void norm_swizzle_kernel(
        const float* __restrict__ x, const float* __restrict__ xa,
        unsigned short* __restrict__ swz, float* __restrict__ posBuckets) {
    const int wid  = (int)((blockIdx.x * blockDim.x + threadIdx.x) >> 5);
    const int lane = (int)(threadIdx.x & 31);
    if (wid >= B_ROWS) return;

    const float4 vx = ((const float4*)(x  + (size_t)wid * DIM))[lane];
    const float4 va = ((const float4*)(xa + (size_t)wid * DIM))[lane];
    float sx = vx.x * vx.x + vx.y * vx.y + vx.z * vx.z + vx.w * vx.w;
    float sa = va.x * va.x + va.y * va.y + va.z * va.z + va.w * va.w;
    float dp = vx.x * va.x + vx.y * va.y + vx.z * va.z + vx.w * va.w;
    #pragma unroll
    for (int off = 16; off; off >>= 1) {
        sx += __shfl_xor(sx, off);
        sa += __shfl_xor(sa, off);
        dp += __shfl_xor(dp, off);
    }
    const float rsx = rsqrtf(sx), rsa = rsqrtf(sa);

    store_row_swizzled(swz, wid,          lane, vx, rsx);
    store_row_swizzled(swz, B_ROWS + wid, lane, va, rsa);

    if (lane == 0)
        atomicAdd(&posBuckets[(wid & (NBUCKET - 1)) * BSTRIDE],
                  dp * rsx * rsa * INV_T);
}

// bits[i] = OR_k (1 << nq[i][k]);  hq[i] copied to int32 array.
__global__ __launch_bounds__(256) void prep_mask_kernel(
        const int* __restrict__ hq_in, const int* __restrict__ nq_in,
        unsigned* __restrict__ bits, int* __restrict__ hq) {
    const int i = (int)(blockIdx.x * blockDim.x + threadIdx.x);
    if (i >= NROWS) return;
    unsigned b = 0;
    #pragma unroll
    for (int k = 0; k < 10; ++k) b |= 1u << (unsigned)nq_in[i * 10 + k];
    bits[i] = b;
    hq[i]   = hq_in[i];
}

// 128x128 output tile per 256-thread block (8 waves). Wave w: 16-row strip,
// 8 column tiles, K=128 in 4 chunks of 32. B-side fragments staged in LDS
// (32 KB, linear coalesced copy thanks to the swizzled layout).
__global__ __launch_bounds__(256) void sim_reduce_kernel(
        const unsigned short* __restrict__ swz,
        const unsigned* __restrict__ bits, const int* __restrict__ hq,
        float* __restrict__ denomBuckets) {
    __shared__ uint4    Bsh[2048];   // 8 tiles * 4 chunks * 64 uint4 = 32 KB
    __shared__ unsigned bitsSh[128];
    __shared__ int      hqSh[128];
    __shared__ float    waveSum[8];

    const int tid  = (int)threadIdx.x;
    const int w    = tid >> 5;
    const int lane = tid & 31;
    const int bRow = (int)blockIdx.y;
    const int bCol = (int)blockIdx.x;

    // Stage column-side fragments: contiguous 32 KB region in swizzled buffer.
    const uint4* src = ((const uint4*)swz) + (size_t)bCol * 2048;
    #pragma unroll
    for (int r = 0; r < 8; ++r) Bsh[r * 256 + tid] = src[r * 256 + tid];
    if (tid < 128) {
        bitsSh[tid] = bits[bRow * 128 + tid];
        hqSh[tid]   = hq[bCol * 128 + tid];
    }
    __syncthreads();

    const int rowTile = bRow * 8 + w;
    const uint4* Abase = ((const uint4*)swz) + (size_t)rowTile * 256;

    v8f acc[8] = {};
    #pragma unroll
    for (int c = 0; c < 4; ++c) {
        union { uint4 u[2]; v16h h; } A;
        A.u[0] = Abase[c * 64 + lane];
        A.u[1] = Abase[c * 64 + 32 + lane];
        #pragma unroll
        for (int t = 0; t < 8; ++t) {
            union { uint4 u[2]; v16h h; } Bf;
            Bf.u[0] = Bsh[(t * 4 + c) * 64 + lane];
            Bf.u[1] = Bsh[(t * 4 + c) * 64 + 32 + lane];
            acc[t] = __builtin_amdgcn_wmma_f32_16x16x32_f16(
                false, A.h, false, Bf.h, (short)0, acc[t], false, false);
        }
    }

    // Fused epilogue: D layout -> row M = r + 8*(lane>>4), col N = lane&15.
    const int half = lane >> 4;
    const int nl   = lane & 15;
    float lsum = 0.0f;
    #pragma unroll
    for (int t = 0; t < 8; ++t) {
        const int jloc = t * 16 + nl;
        const int j    = bCol * 128 + jloc;
        const int hqj  = hqSh[jloc];
        #pragma unroll
        for (int r = 0; r < 8; ++r) {
            const int iloc = w * 16 + half * 8 + r;
            const int i    = bRow * 128 + iloc;
            const unsigned bi = bitsSh[iloc];
            if (((bi >> hqj) & 1u) && (i != j))
                lsum += __expf(acc[t][r] * INV_T);
        }
    }
    // Wave reduce, then block reduce in LDS -> single spread atomic per block.
    #pragma unroll
    for (int off = 16; off; off >>= 1) lsum += __shfl_xor(lsum, off);
    if (lane == 0) waveSum[w] = lsum;
    __syncthreads();
    if (tid == 0) {
        float bsum = 0.0f;
        #pragma unroll
        for (int k = 0; k < 8; ++k) bsum += waveSum[k];
        const int bucket = (bRow * 131 + bCol) & (NBUCKET - 1);
        atomicAdd(&denomBuckets[bucket * BSTRIDE], bsum);
    }
}

__global__ void finalize_kernel(const float* __restrict__ denomBuckets,
                                const float* __restrict__ posBuckets,
                                float* __restrict__ out) {
    float denom = 0.0f, possum = 0.0f;
    #pragma unroll
    for (int k = 0; k < NBUCKET; ++k) {
        denom  += denomBuckets[k * BSTRIDE];
        possum += posBuckets[k * BSTRIDE];
    }
    // loss = log(denom) - (2 * possum) / N = log(denom) - possum / B
    out[0] = logf(denom) - possum * (1.0f / (float)B_ROWS);
}

extern "C" void kernel_launch(void* const* d_in, const int* in_sizes, int n_in,
                              void* d_out, int out_size, void* d_ws, size_t ws_size,
                              hipStream_t stream) {
    const float* x     = (const float*)d_in[0];
    const float* xa    = (const float*)d_in[1];
    const int*   hq_in = (const int*)d_in[2];
    const int*   nq_in = (const int*)d_in[3];

    char* ws = (char*)d_ws;
    float*          denomB = (float*)ws;                         // 8 KB
    float*          posB   = (float*)(ws + 8192);                // 8 KB
    unsigned short* swz    = (unsigned short*)(ws + 16384);
    const size_t swzBytes  = (size_t)NROWS * DIM * 2;            // 4 MB
    unsigned*       bits   = (unsigned*)(ws + 16384 + swzBytes);
    int*            hq     = (int*)(ws + 16384 + swzBytes + (size_t)NROWS * 4);

    hipMemsetAsync(ws, 0, 16384, stream);
    norm_swizzle_kernel<<<dim3(B_ROWS * 32 / 256), 256, 0, stream>>>(x, xa, swz, posB);
    prep_mask_kernel<<<dim3(NROWS / 256), 256, 0, stream>>>(hq_in, nq_in, bits, hq);
    sim_reduce_kernel<<<dim3(128, 128), 256, 0, stream>>>(swz, bits, hq, denomB);
    finalize_kernel<<<1, 1, 0, stream>>>(denomB, posB, (float*)d_out);
}
